// AdjustLayer_7095285973044
// MI455X (gfx1250) — matrix operand order repaired and checked by reading, not compiled, not run on hardware
//
#include <hip/hip_runtime.h>
#include <hip/hip_bf16.h>
#include <stdint.h>

typedef __attribute__((ext_vector_type(16))) __bf16 v16bf;
typedef __attribute__((ext_vector_type(8)))  __bf16 v8bf;
typedef __attribute__((ext_vector_type(8)))  float  v8f;

#define B_     64
#define CIN    1024
#define COUT   256
#define HH     36
#define WW     36
#define HW     (HH*WW)          // 1296
#define HP     38
#define HPW    (HP*HP)          // 1444 padded plane
#define PH     18
#define PW     18

// ---------------- workspace layout (bytes) ----------------
#define OFF_H0P   0ULL          // bf16 [64][256][38][38] : 47,316,992
#define OFF_W1P   47316992ULL   // bf16 [256][1024]       :    524,288
#define OFF_W2P   47841280ULL   // bf16 [9][256][256]     :  1,179,648
#define OFF_B1    49020928ULL   // f32  [256]
#define OFF_B2    49021952ULL   // f32  [256]

// ------------------------------------------------------------------
// Tensor Data Mover: load a 2D bf16 tile (tile_w x tile_h, row pitch
// stride_elems) from global into LDS at lds_addr. D# per ISA ch.8.
// This toolchain exposes the 6-arg builtin:
//   (u32x4 g0, i32x8 g1, i32x4 g2, i32x4 g3, i32x8 pad, i32 cpol)
#if __has_builtin(__builtin_amdgcn_tensor_load_to_lds)
#define USE_TDM 1
typedef unsigned int tdm_u4 __attribute__((ext_vector_type(4)));
typedef int          tdm_i8 __attribute__((ext_vector_type(8)));
typedef int          tdm_i4 __attribute__((ext_vector_type(4)));

__device__ __forceinline__ void tdm_load_2d(unsigned lds_addr, const void* gptr,
                                            unsigned tile_w, unsigned tile_h,
                                            unsigned stride_elems) {
    unsigned long long ga = (unsigned long long)(uintptr_t)gptr;
    // group0: [1:0]=count=1 | [63:32]=lds_addr | [120:64]=global_addr | [127:126]=type2
    tdm_u4 g0 = { 1u,
                  lds_addr,
                  (unsigned)ga,
                  (unsigned)((ga >> 32) & 0x1FFFFFFu) | (2u << 30) };
    // group1: data_size=1 (2B); tensor_dim0/1 = tile dims; tile_dim0/1; dim0 stride
    tdm_i8 g1 = { (int)0x00010000,                            // data_size=2B
                  (int)(tile_w << 16),                        // tensor_dim0[15:0]
                  (int)((tile_w >> 16) | (tile_h << 16)),     // dim0 hi | dim1 lo
                  (int)(tile_w << 16),                        // dim1 hi=0 | tile_dim0
                  (int)(tile_h & 0xFFFF),                     // tile_dim1 | tile_dim2=0
                  (int)stride_elems,                          // dim0 stride lo32
                  0, 0 };
    tdm_i4 gz4 = { 0, 0, 0, 0 };
    tdm_i8 gz8 = { 0, 0, 0, 0, 0, 0, 0, 0 };
    __builtin_amdgcn_tensor_load_to_lds(g0, g1, gz4, gz4, gz8, 0);
}
#else
#define USE_TDM 0
#endif

// ------------------------------------------------------------------
__global__ void zero_u4(uint4* p, long n) {
    long i = (long)blockIdx.x * blockDim.x + threadIdx.x;
    long stride = (long)gridDim.x * blockDim.x;
    uint4 z; z.x = 0u; z.y = 0u; z.z = 0u; z.w = 0u;
    for (; i < n; i += stride) p[i] = z;
}

__global__ void prep_w1(const float* __restrict__ w1, const float* __restrict__ g1,
                        const float* __restrict__ b1, const float* __restrict__ m1,
                        const float* __restrict__ v1, __bf16* __restrict__ w1p,
                        float* __restrict__ bias1) {
    int i = blockIdx.x * blockDim.x + threadIdx.x;
    if (i >= COUT * CIN) return;
    int co = i >> 10;
    float s = g1[co] * rsqrtf(v1[co] + 1e-5f);
    w1p[i] = (__bf16)(w1[i] * s);
    if ((i & 1023) == 0) bias1[co] = b1[co] - m1[co] * s;
}

__global__ void prep_w2(const float* __restrict__ w2, const float* __restrict__ g2,
                        const float* __restrict__ b2, const float* __restrict__ m2,
                        const float* __restrict__ v2, __bf16* __restrict__ w2p,
                        float* __restrict__ bias2) {
    int i = blockIdx.x * blockDim.x + threadIdx.x;
    if (i >= 9 * COUT * COUT) return;
    int ci = i & 255;
    int co = (i >> 8) & 255;
    int s  = i >> 16;                 // 0..8 = dy*3+dx
    int dy = s / 3, dx = s % 3;
    float sc = g2[co] * rsqrtf(v2[co] + 1e-5f);
    w2p[i] = (__bf16)(w2[((co * 256 + ci) * 3 + dy) * 3 + dx] * sc);
    if (i < COUT) bias2[i] = b2[i] - m2[i] * g2[i] * rsqrtf(v2[i] + 1e-5f);
}

// ------------------------------------------------------------------
// conv1x1 + folded BN1 -> h0p (bf16, zero-padded planes)
// grid (21, 2, 64), block 256.  M=128 co, N=64 px, K staged as two
// TDM-resident 128x512 bf16 A-panels; B transposed+cvt through LDS.
__global__ __launch_bounds__(256)
void conv1_wmma(const float* __restrict__ xin, const __bf16* __restrict__ w1p,
                const float* __restrict__ bias1, __bf16* __restrict__ h0p) {
    __shared__ __align__(16) __bf16 Ap[128][512];   // 128 KB weight panel
    __shared__ __align__(16) __bf16 Bs[64][32];     // [pixel][k]
    const int tid   = threadIdx.x;
    const int nbase = blockIdx.x * 64;
    const int mbase = blockIdx.y * 128;
    const int b     = blockIdx.z;
    const int lane  = tid & 31;
    const int wave  = tid >> 5;
    const int wm    = wave >> 1;          // 0..3
    const int wn    = wave & 1;           // 0..1
    v8f acc[2][2] = {};

    const int ncol = tid & 63;
    const int krb2 = (tid >> 6) * 2;      // 0,2,4,6
    int nclamp = nbase + ncol; if (nclamp > HW - 1) nclamp = HW - 1;
    const float* xbase = xin + (size_t)b * CIN * HW + nclamp;

    for (int half = 0; half < 2; ++half) {
#if USE_TDM
        if (wave == 0) {
            tdm_load_2d((unsigned)(uintptr_t)&Ap[0][0],
                        w1p + (size_t)mbase * CIN + half * 512,
                        512u, 128u, (unsigned)CIN);
            __builtin_amdgcn_s_wait_tensorcnt(0);
        }
#else
        for (int ch = tid; ch < 128 * 512 / 8; ch += 256) {
            int row = ch >> 6, cc = (ch & 63) * 8;
            *(uint4*)&Ap[row][cc] =
                *(const uint4*)&w1p[(size_t)(mbase + row) * CIN + half * 512 + cc];
        }
#endif
        __syncthreads();
        for (int kt = 0; kt < 16; ++kt) {
            const int k0 = kt * 32;
            const int kg = half * 512 + k0;
            // stage B: coalesced f32 loads, packed bf16 pair stores
#pragma unroll
            for (int pass = 0; pass < 4; ++pass) {
                int kr = krb2 + pass * 8;
                float v0 = xbase[(size_t)(kg + kr) * HW];
                float v1 = xbase[(size_t)(kg + kr + 1) * HW];
                union { __bf16 h[2]; unsigned u; } pk;
                pk.h[0] = (__bf16)v0; pk.h[1] = (__bf16)v1;
                *(unsigned*)&Bs[ncol][kr] = pk.u;
            }
            __syncthreads();
            const int khi = lane >> 4;
            const int kseg = khi * 16;
            union { v16bf v; v8bf h[2]; } a0, a1, bb0, bb1;
            const int r0 = wm * 32 + (lane & 15);
            a0.h[0] = *(const v8bf*)&Ap[r0][k0 + 8 * khi];
            a0.h[1] = *(const v8bf*)&Ap[r0][k0 + 16 + 8 * khi];
            a1.h[0] = *(const v8bf*)&Ap[r0 + 16][k0 + 8 * khi];
            a1.h[1] = *(const v8bf*)&Ap[r0 + 16][k0 + 16 + 8 * khi];
            const int c0 = wn * 32 + (lane & 15);
            bb0.h[0] = *(const v8bf*)&Bs[c0][8 * khi];
            bb0.h[1] = *(const v8bf*)&Bs[c0][16 + 8 * khi];
            bb1.h[0] = *(const v8bf*)&Bs[c0 + 16][8 * khi];
            bb1.h[1] = *(const v8bf*)&Bs[c0 + 16][16 + 8 * khi];
            acc[0][0] = __builtin_amdgcn_wmma_f32_16x16x32_bf16(false, a0.v, false, bb0.v, (short)0, acc[0][0], false, false);
            acc[0][1] = __builtin_amdgcn_wmma_f32_16x16x32_bf16(false, a0.v, false, bb1.v, (short)0, acc[0][1], false, false);
            acc[1][0] = __builtin_amdgcn_wmma_f32_16x16x32_bf16(false, a1.v, false, bb0.v, (short)0, acc[1][0], false, false);
            acc[1][1] = __builtin_amdgcn_wmma_f32_16x16x32_bf16(false, a1.v, false, bb1.v, (short)0, acc[1][1], false, false);
            __syncthreads();
        }
    }
#pragma unroll
    for (int mi = 0; mi < 2; ++mi)
#pragma unroll
        for (int ni = 0; ni < 2; ++ni)
#pragma unroll
            for (int r = 0; r < 8; ++r) {
                int co = mbase + wm * 32 + mi * 16 + ((lane & 16) ? r + 8 : r);
                int n  = nbase + wn * 32 + ni * 16 + (lane & 15);
                if (n < HW) {
                    int yy = n / WW, xx = n % WW;
                    float v = acc[mi][ni][r] + bias1[co];
                    h0p[((size_t)(b * COUT + co)) * HPW + (yy + 1) * HP + (xx + 1)] = (__bf16)v;
                }
            }
}

// ------------------------------------------------------------------
// conv3x3 (9 shifted 1x1 GEMMs over padded bf16 h0) + BN2 + ReLU -> h f32
// grid (21, 2, 64), block 256.  Per shift: one TDM 128x256 weight panel.
__global__ __launch_bounds__(256)
void conv2_wmma(const __bf16* __restrict__ h0p, const __bf16* __restrict__ w2p,
                const float* __restrict__ bias2, float* __restrict__ hout) {
    __shared__ __align__(16) __bf16 Wp[128][256];   // 64 KB weight panel
    __shared__ __align__(16) __bf16 Bs[64][32];
    const int tid   = threadIdx.x;
    const int nbase = blockIdx.x * 64;
    const int mbase = blockIdx.y * 128;
    const int b     = blockIdx.z;
    const int lane  = tid & 31;
    const int wave  = tid >> 5;
    const int wm    = wave >> 1;
    const int wn    = wave & 1;
    v8f acc[2][2] = {};

    const int ncol = tid & 63;
    const int krb2 = (tid >> 6) * 2;
    int nclamp = nbase + ncol; if (nclamp > HW - 1) nclamp = HW - 1;
    const int ny = nclamp / WW, nx = nclamp % WW;

    for (int s = 0; s < 9; ++s) {
        const int dy = s / 3, dx = s % 3;
#if USE_TDM
        if (wave == 0) {
            tdm_load_2d((unsigned)(uintptr_t)&Wp[0][0],
                        w2p + (size_t)s * (COUT * COUT) + (size_t)mbase * COUT,
                        256u, 128u, (unsigned)COUT);
            __builtin_amdgcn_s_wait_tensorcnt(0);
        }
#else
        for (int ch = tid; ch < 128 * 256 / 8; ch += 256) {
            int row = ch >> 5, cc = (ch & 31) * 8;
            *(uint4*)&Wp[row][cc] =
                *(const uint4*)&w2p[(size_t)s * (COUT * COUT) + (size_t)(mbase + row) * COUT + cc];
        }
#endif
        __syncthreads();
        const __bf16* hbase =
            h0p + (size_t)b * COUT * HPW + (size_t)(ny + dy) * HP + (nx + dx);
        for (int kb = 0; kb < 8; ++kb) {
            const int k0 = kb * 32;
#pragma unroll
            for (int pass = 0; pass < 4; ++pass) {
                int kr = krb2 + pass * 8;
                union { __bf16 h[2]; unsigned u; } pk;
                pk.h[0] = hbase[(size_t)(k0 + kr) * HPW];
                pk.h[1] = hbase[(size_t)(k0 + kr + 1) * HPW];
                *(unsigned*)&Bs[ncol][kr] = pk.u;
            }
            __syncthreads();
            const int khi = lane >> 4;
            const int kseg = khi * 16;
            union { v16bf v; v8bf h[2]; } a0, a1, bb0, bb1;
            const int r0 = wm * 32 + (lane & 15);
            a0.h[0] = *(const v8bf*)&Wp[r0][k0 + 8 * khi];
            a0.h[1] = *(const v8bf*)&Wp[r0][k0 + 16 + 8 * khi];
            a1.h[0] = *(const v8bf*)&Wp[r0 + 16][k0 + 8 * khi];
            a1.h[1] = *(const v8bf*)&Wp[r0 + 16][k0 + 16 + 8 * khi];
            const int c0 = wn * 32 + (lane & 15);
            bb0.h[0] = *(const v8bf*)&Bs[c0][8 * khi];
            bb0.h[1] = *(const v8bf*)&Bs[c0][16 + 8 * khi];
            bb1.h[0] = *(const v8bf*)&Bs[c0 + 16][8 * khi];
            bb1.h[1] = *(const v8bf*)&Bs[c0 + 16][16 + 8 * khi];
            acc[0][0] = __builtin_amdgcn_wmma_f32_16x16x32_bf16(false, a0.v, false, bb0.v, (short)0, acc[0][0], false, false);
            acc[0][1] = __builtin_amdgcn_wmma_f32_16x16x32_bf16(false, a0.v, false, bb1.v, (short)0, acc[0][1], false, false);
            acc[1][0] = __builtin_amdgcn_wmma_f32_16x16x32_bf16(false, a1.v, false, bb0.v, (short)0, acc[1][0], false, false);
            acc[1][1] = __builtin_amdgcn_wmma_f32_16x16x32_bf16(false, a1.v, false, bb1.v, (short)0, acc[1][1], false, false);
            __syncthreads();
        }
    }
#pragma unroll
    for (int mi = 0; mi < 2; ++mi)
#pragma unroll
        for (int ni = 0; ni < 2; ++ni)
#pragma unroll
            for (int r = 0; r < 8; ++r) {
                int co = mbase + wm * 32 + mi * 16 + ((lane & 16) ? r + 8 : r);
                int n  = nbase + wn * 32 + ni * 16 + (lane & 15);
                if (n < HW) {
                    float v = acc[mi][ni][r] + bias2[co];
                    v = fmaxf(v, 0.0f);
                    hout[((size_t)(b * COUT + co)) * HW + n] = v;
                }
            }
}

// ------------------------------------------------------------------
__device__ __forceinline__ float hatF(float t) {
    t = fminf(1.0f, fmaxf(-1.0f, t));
    return t - copysignf(0.5f * t * t, t);
}
__device__ __forceinline__ float hat_int(float a, float bb, float p) {
    return hatF(bb - p) - hatF(a - p);
}

__global__ __launch_bounds__(256)
void prroi_kernel(const float* __restrict__ h, const float* __restrict__ roi,
                  float* __restrict__ rf, float* __restrict__ roi5) {
    __shared__ float wy[PH][HH];
    __shared__ float wx[PW][WW];
    __shared__ float tmp[PH * WW];
    __shared__ float ftile[HW];
    const int n   = blockIdx.x;
    const int tid = threadIdx.x;

    float rx = roi[n * 4 + 0], ry = roi[n * 4 + 1];
    float rw = roi[n * 4 + 2], rh = roi[n * 4 + 3];
    if (tid == 0) {
        roi5[n * 5 + 0] = (float)n;
        roi5[n * 5 + 1] = rx;  roi5[n * 5 + 2] = ry;
        roi5[n * 5 + 3] = rx + rw;  roi5[n * 5 + 4] = ry + rh;
    }
    const float sc = 0.125f;
    float x1 = rx * sc, y1 = ry * sc;
    float x2 = (rx + rw) * sc, y2 = (ry + rh) * sc;
    float bw = (x2 - x1) / (float)PW;
    float bh = (y2 - y1) / (float)PH;
    for (int e = tid; e < PH * HH; e += 256) {
        int p = e / HH, g = e % HH;
        float ya = y1 + bh * p;
        wy[p][g] = hat_int(ya, ya + bh, (float)g);
        float xa = x1 + bw * p;
        wx[p][g] = hat_int(xa, xa + bw, (float)g);
    }
    __syncthreads();
    float area = bw * bh;
    float inv  = (area > 0.0f) ? 1.0f / fmaxf(area, 1e-12f) : 0.0f;

    for (int c = 0; c < COUT; ++c) {
        const float* f = h + ((size_t)n * COUT + c) * HW;
        for (int e = tid; e < HW; e += 256) ftile[e] = f[e];
        __syncthreads();
        for (int e = tid; e < PH * WW; e += 256) {
            int p = e / WW, w = e % WW;
            float s = 0.0f;
#pragma unroll 4
            for (int y = 0; y < HH; ++y) s = fmaf(wy[p][y], ftile[y * WW + w], s);
            tmp[e] = s;
        }
        __syncthreads();
        for (int e = tid; e < PH * PW; e += 256) {
            int p = e / PW, q = e % PW;
            float s = 0.0f;
#pragma unroll 4
            for (int w = 0; w < WW; ++w) s = fmaf(tmp[p * WW + w], wx[q][w], s);
            rf[(((size_t)n * COUT + c) * PH + p) * PW + q] = s * inv;
        }
        __syncthreads();
    }
}

// ------------------------------------------------------------------
extern "C" void kernel_launch(void* const* d_in, const int* in_sizes, int n_in,
                              void* d_out, int out_size, void* d_ws, size_t ws_size,
                              hipStream_t stream) {
    const float* x   = (const float*)d_in[0];
    const float* roi = (const float*)d_in[1];
    const float* w1  = (const float*)d_in[2];
    const float* g1  = (const float*)d_in[3];
    const float* b1  = (const float*)d_in[4];
    const float* m1  = (const float*)d_in[5];
    const float* v1  = (const float*)d_in[6];
    const float* w2  = (const float*)d_in[7];
    const float* g2  = (const float*)d_in[8];
    const float* b2  = (const float*)d_in[9];
    const float* m2  = (const float*)d_in[10];
    const float* v2  = (const float*)d_in[11];

    char* ws = (char*)d_ws;
    __bf16* h0p   = (__bf16*)(ws + OFF_H0P);
    __bf16* w1p   = (__bf16*)(ws + OFF_W1P);
    __bf16* w2p   = (__bf16*)(ws + OFF_W2P);
    float*  bias1 = (float*)(ws + OFF_B1);
    float*  bias2 = (float*)(ws + OFF_B2);

    float* hout    = (float*)d_out;                       // [64][256][36][36]
    float* rfout   = hout + (size_t)B_ * COUT * HW;       // [64][256][18][18]
    float* roi5out = rfout + (size_t)B_ * COUT * PH * PW; // [64][5]

    long n16 = (long)(OFF_W1P / 16);
    zero_u4<<<4096, 256, 0, stream>>>((uint4*)h0p, n16);
    prep_w1<<<(COUT * CIN + 255) / 256, 256, 0, stream>>>(w1, g1, b1, m1, v1, w1p, bias1);
    prep_w2<<<(9 * COUT * COUT + 255) / 256, 256, 0, stream>>>(w2, g2, b2, m2, v2, w2p, bias2);
    dim3 cgrid(21, 2, B_);
    conv1_wmma<<<cgrid, 256, 0, stream>>>(x, w1p, bias1, h0p);
    conv2_wmma<<<cgrid, 256, 0, stream>>>(h0p, w2p, bias2, hout);
    prroi_kernel<<<B_, 256, 0, stream>>>(hout, roi, rfout, roi5out);
}
